// MambaBlock_90460601189093
// MI455X (gfx1250) — matrix operand order repaired and checked by reading, not compiled, not run on hardware
//
#include <hip/hip_runtime.h>

// ---------------- problem constants ----------------
#define T_TOK  32768   // B * L
#define LSEQ   4096
#define BATCH  8
#define DMODEL 256
#define DINX   512     // expanded dim
#define NSTATE 16
#define HIDN   1024

typedef __attribute__((ext_vector_type(16))) __bf16 v16bf;
typedef __attribute__((ext_vector_type(8)))  __bf16 v8bf;
typedef __attribute__((ext_vector_type(8)))  float  v8f;

union ABv { v16bf v; v8bf h[2]; };

__device__ __forceinline__ float sigm(float x) { return 1.0f / (1.0f + __expf(-x)); }

// ---------------- fp32 -> bf16 convert (weights) ----------------
__global__ void k_cvt_bf16(const float* __restrict__ s, __bf16* __restrict__ d, int n) {
  int i = blockIdx.x * 256 + threadIdx.x;
  if (i < n) d[i] = (__bf16)s[i];
}

// ---------------- RMSNorm: one 256-wide row per block, emit bf16 ----------------
__global__ void k_rmsnorm(const float* __restrict__ x, const float* __restrict__ w,
                          __bf16* __restrict__ out) {
  __shared__ float red[256];
  const int row = blockIdx.x;
  const int t = threadIdx.x;
  float v = x[(size_t)row * DMODEL + t];
  red[t] = v * v;
  __syncthreads();
  for (int s = 128; s > 0; s >>= 1) {
    if (t < s) red[t] += red[t + s];
    __syncthreads();
  }
  float rms = sqrtf(red[0] * (1.0f / DMODEL) + 1e-5f);
  out[(size_t)row * DMODEL + t] = (__bf16)(v / rms * w[t]);
}

// ---------------- WMMA GEMM core: out = A[T,K] x W[N,K]^T ----------------
// Per wave: MT*16 rows x NT*16 cols; A fragment reused across NT tiles,
// B fragment reused across MT tiles (8 WMMAs per 12 b128 loads at MT=2,NT=4).
template <int K, int MT, int NT>
__device__ __forceinline__ void gemm_tiles(const __bf16* __restrict__ A,
                                           const __bf16* __restrict__ W,
                                           int rowBase, int colBase, v8f acc[MT][NT]) {
  const int lane = threadIdx.x & 31;
  const int m  = lane & 15;     // A row within tile / B column within tile
  const int hi = lane >> 4;     // K-half select per ISA layout
  const __bf16* arow[MT];
#pragma unroll
  for (int mt = 0; mt < MT; ++mt)
    arow[mt] = A + (size_t)(rowBase + mt * 16 + m) * K;
  for (int kk = 0; kk < K; kk += 32) {
    ABv a[MT];
#pragma unroll
    for (int mt = 0; mt < MT; ++mt) {
      a[mt].h[0] = *(const v8bf*)(arow[mt] + kk + hi * 8);      // K = kk+[0..7]/[8..15]
      a[mt].h[1] = *(const v8bf*)(arow[mt] + kk + 16 + hi * 8); // K = kk+[16..23]/[24..31]
    }
#pragma unroll
    for (int t = 0; t < NT; ++t) {
      const __bf16* wrow = W + (size_t)(colBase + t * 16 + m) * K + kk + hi * 8;
      ABv b;
      b.h[0] = *(const v8bf*)(wrow);      // B col = lane, K = kk + hi*8 + [0..7]
      b.h[1] = *(const v8bf*)(wrow + 16); //                K = kk + 16 + hi*8 + [0..7]
#pragma unroll
      for (int mt = 0; mt < MT; ++mt)
        acc[mt][t] = __builtin_amdgcn_wmma_f32_16x16x32_bf16(
            false, a[mt].v, false, b.v, (short)0, acc[mt][t], false, false);
    }
  }
}

// ---------------- in_proj: xz = xn x Win^T ; split into u / z (fp32) ----------------
// grid (T/32, 2), block 256 (8 waves x 64 cols, 32 rows)
__global__ void k_gemm_inproj(const __bf16* __restrict__ A, const __bf16* __restrict__ W,
                              float* __restrict__ u, float* __restrict__ z) {
  const int wave = threadIdx.x >> 5;
  const int rowBase = blockIdx.x * 32;
  const int colBase = blockIdx.y * 512 + wave * 64;
  v8f acc[2][4] = {};
  gemm_tiles<DMODEL, 2, 4>(A, W, rowBase, colBase, acc);
  const int lane = threadIdx.x & 31;
  const int n = lane & 15, hi = lane >> 4;
#pragma unroll
  for (int mt = 0; mt < 2; ++mt) {
#pragma unroll
    for (int t = 0; t < 4; ++t) {
      int col = colBase + t * 16 + n;
#pragma unroll
      for (int r = 0; r < 8; ++r) {
        int row = rowBase + mt * 16 + hi * 8 + r;
        float v = acc[mt][t][r];
        if (col < DINX) u[(size_t)row * DINX + col] = v;
        else            z[(size_t)row * DINX + (col - DINX)] = v;
      }
    }
  }
}

// ---------------- out_proj: x1 = x0 + ybf x Wout^T ---------------- grid (T/32), block 128
__global__ void k_gemm_outproj(const __bf16* __restrict__ A, const __bf16* __restrict__ W,
                               const float* __restrict__ x0, float* __restrict__ x1) {
  const int wave = threadIdx.x >> 5;
  const int rowBase = blockIdx.x * 32;
  const int colBase = wave * 64;
  v8f acc[2][4] = {};
  gemm_tiles<DINX, 2, 4>(A, W, rowBase, colBase, acc);
  const int lane = threadIdx.x & 31;
  const int n = lane & 15, hi = lane >> 4;
#pragma unroll
  for (int mt = 0; mt < 2; ++mt) {
#pragma unroll
    for (int t = 0; t < 4; ++t) {
      int col = colBase + t * 16 + n;
#pragma unroll
      for (int r = 0; r < 8; ++r) {
        size_t o = (size_t)(rowBase + mt * 16 + hi * 8 + r) * DMODEL + col;
        x1[o] = x0[o] + acc[mt][t][r];
      }
    }
  }
}

// ---------------- fc1 (paired columns): m = silu(u)*v, emit bf16 ----------------
// grid (T/16, 2), block 256; col in [0,1024); u = cols, v = cols+1024 of fc1_w
// MT=1 here: the u/v pairing already needs 8 accumulators; MT=2 would risk spills.
__global__ void k_gemm_fc1(const __bf16* __restrict__ A, const __bf16* __restrict__ W,
                           const float* __restrict__ bias, __bf16* __restrict__ mout) {
  const int wave = threadIdx.x >> 5;
  const int rowBase = blockIdx.x * 16;
  const int colBase = blockIdx.y * 512 + wave * 64;
  v8f au[1][4] = {}, av[1][4] = {};
  gemm_tiles<DMODEL, 1, 4>(A, W, rowBase, colBase, au);
  gemm_tiles<DMODEL, 1, 4>(A, W + (size_t)HIDN * DMODEL, rowBase, colBase, av);
  const int lane = threadIdx.x & 31;
  const int n = lane & 15, hi = lane >> 4;
#pragma unroll
  for (int t = 0; t < 4; ++t) {
    int col = colBase + t * 16 + n;
#pragma unroll
    for (int r = 0; r < 8; ++r) {
      int row = rowBase + hi * 8 + r;
      float uu = au[0][t][r] + bias[col];
      float vv = av[0][t][r] + bias[HIDN + col];
      mout[(size_t)row * HIDN + col] = (__bf16)(uu * sigm(uu) * vv);
    }
  }
}

// ---------------- fc2: out = x1 + m x Wfc2^T + b ---------------- grid (T/32), block 128
__global__ void k_gemm_fc2(const __bf16* __restrict__ A, const __bf16* __restrict__ W,
                           const float* __restrict__ bias, const float* __restrict__ x1,
                           float* __restrict__ out) {
  const int wave = threadIdx.x >> 5;
  const int rowBase = blockIdx.x * 32;
  const int colBase = wave * 64;
  v8f acc[2][4] = {};
  gemm_tiles<HIDN, 2, 4>(A, W, rowBase, colBase, acc);
  const int lane = threadIdx.x & 31;
  const int n = lane & 15, hi = lane >> 4;
#pragma unroll
  for (int mt = 0; mt < 2; ++mt) {
#pragma unroll
    for (int t = 0; t < 4; ++t) {
      int col = colBase + t * 16 + n;
#pragma unroll
      for (int r = 0; r < 8; ++r) {
        size_t o = (size_t)(rowBase + mt * 16 + hi * 8 + r) * DMODEL + col;
        out[o] = x1[o] + acc[mt][t][r] + bias[col];
      }
    }
  }
}

// ---------------- depthwise causal conv (K=4) + SiLU ----------------
__global__ void k_conv(const float* __restrict__ u, const float* __restrict__ cw,
                       const float* __restrict__ cb, float* __restrict__ uc) {
  size_t i = (size_t)blockIdx.x * 256 + threadIdx.x;   // over T*DINX
  if (i >= (size_t)T_TOK * DINX) return;
  int d = (int)(i & (DINX - 1));
  size_t t = i >> 9;                                   // token index b*L + l
  int l = (int)(t & (LSEQ - 1));
  float s = cb[d];
#pragma unroll
  for (int j = 0; j < 4; ++j) {
    int ll = l - 3 + j;
    if (ll >= 0) s += cw[d * 4 + j] * u[(t - 3 + j) * DINX + d];
  }
  uc[i] = s * sigm(s);
}

// ---------------- x_proj: [T,512] -> dt[16], B[16], C[16] ----------------
__global__ void k_xproj(const float* __restrict__ uc, const float* __restrict__ w,
                        float* __restrict__ dtv, float* __restrict__ Bv,
                        float* __restrict__ Cv) {
  int idx = blockIdx.x * 256 + threadIdx.x;            // over T*48
  if (idx >= T_TOK * 48) return;
  int j = idx % 48;
  int t = idx / 48;
  const float* ur = uc + (size_t)t * DINX;
  const float* wr = w + (size_t)j * DINX;
  float s = 0.f;
  for (int k = 0; k < DINX; k += 4) {
    s += ur[k] * wr[k] + ur[k + 1] * wr[k + 1] + ur[k + 2] * wr[k + 2] + ur[k + 3] * wr[k + 3];
  }
  if      (j < 16) dtv[(size_t)t * 16 + j]      = s;
  else if (j < 32) Bv [(size_t)t * 16 + j - 16] = s;
  else             Cv [(size_t)t * 16 + j - 32] = s;
}

// ---------------- dt_proj + softplus ----------------
__global__ void k_dtproj(const float* __restrict__ dtv, const float* __restrict__ w,
                         const float* __restrict__ b, float* __restrict__ delta) {
  size_t i = (size_t)blockIdx.x * 256 + threadIdx.x;   // over T*DINX
  if (i >= (size_t)T_TOK * DINX) return;
  int d = (int)(i & (DINX - 1));
  size_t t = i >> 9;
  float s = b[d];
#pragma unroll
  for (int r = 0; r < 16; ++r) s += dtv[t * 16 + r] * w[d * 16 + r];
  delta[i] = (s > 20.f) ? s : log1pf(__expf(s));
}

// ---------------- selective scan: h[16] in VGPRs, B/C chunked in LDS ----------------
// grid 64 (8 batch x 8 chan-chunks of 64), block 64; fuses D-skip + silu(z) gate, emits bf16
#define SCAN_CH 64
__global__ void k_scan(const float* __restrict__ delta, const float* __restrict__ uc,
                       const float* __restrict__ Bv, const float* __restrict__ Cv,
                       const float* __restrict__ zbuf, const float* __restrict__ Alog,
                       const float* __restrict__ Dsk, __bf16* __restrict__ ybf) {
  __shared__ float sB[SCAN_CH * 16];
  __shared__ float sC[SCAN_CH * 16];
  const int b = blockIdx.x >> 3;
  const int d = ((blockIdx.x & 7) << 6) + threadIdx.x;
  float A2[16], h[16];
#pragma unroll
  for (int n = 0; n < 16; ++n) {
    A2[n] = -__expf(Alog[d * 16 + n]) * 1.4426950408889634f;  // A * log2(e)
    h[n] = 0.f;
  }
  const float dskip = Dsk[d];
  const size_t base = (size_t)b * LSEQ;
  for (int l0 = 0; l0 < LSEQ; l0 += SCAN_CH) {
    __syncthreads();
    for (int i = threadIdx.x; i < SCAN_CH * 16; i += 64) {
      sB[i] = Bv[(base + l0) * 16 + i];
      sC[i] = Cv[(base + l0) * 16 + i];
    }
    __syncthreads();
    for (int li = 0; li < SCAN_CH; ++li) {
      size_t idx = (base + l0 + li) * DINX + d;
      float dl = delta[idx];
      float ut = uc[idx];
      float du = dl * ut;
      float y = 0.f;
      const float* Bl = sB + li * 16;
      const float* Cl = sC + li * 16;
#pragma unroll
      for (int n = 0; n < 16; ++n) {
        h[n] = exp2f(dl * A2[n]) * h[n] + du * Bl[n];   // v_exp_f32 path
        y += h[n] * Cl[n];
      }
      y += ut * dskip;
      float zz = zbuf[idx];
      y *= zz * sigm(zz);
      ybf[idx] = (__bf16)y;
    }
  }
}

// ---------------- host-side launch ----------------
extern "C" void kernel_launch(void* const* d_in, const int* in_sizes, int n_in,
                              void* d_out, int out_size, void* d_ws, size_t ws_size,
                              hipStream_t stream) {
  (void)in_sizes; (void)n_in; (void)out_size; (void)ws_size;
  const float* x      = (const float*)d_in[0];
  const float* n1w    = (const float*)d_in[1];
  const float* inW    = (const float*)d_in[2];
  const float* convW  = (const float*)d_in[3];
  const float* convB  = (const float*)d_in[4];
  const float* xprojW = (const float*)d_in[5];
  const float* dtW    = (const float*)d_in[6];
  const float* dtB    = (const float*)d_in[7];
  const float* Alog   = (const float*)d_in[8];
  const float* Dsk    = (const float*)d_in[9];
  const float* outW   = (const float*)d_in[10];
  const float* n2w    = (const float*)d_in[11];
  const float* fc1W   = (const float*)d_in[12];
  const float* fc1B   = (const float*)d_in[13];
  const float* fc2W   = (const float*)d_in[14];
  const float* fc2B   = (const float*)d_in[15];
  float* out = (float*)d_out;

  char* p = (char*)d_ws;
  auto alloc = [&](size_t bytes) -> char* {
    char* r = p;
    p += (bytes + 255) & ~(size_t)255;
    return r;
  };
  const size_t T = T_TOK;
  __bf16* wInB  = (__bf16*)alloc((size_t)2 * DINX * DMODEL * 2);
  __bf16* wOutB = (__bf16*)alloc((size_t)DMODEL * DINX * 2);
  __bf16* wFc1B = (__bf16*)alloc((size_t)2 * HIDN * DMODEL * 2);
  __bf16* wFc2B = (__bf16*)alloc((size_t)DMODEL * HIDN * 2);
  __bf16* xnB   = (__bf16*)alloc(T * DMODEL * 2);
  float*  ubuf  = (float*) alloc(T * DINX * 4);
  float*  zbuf  = (float*) alloc(T * DINX * 4);
  float*  ucbuf = (float*) alloc(T * DINX * 4);
  float*  dtv   = (float*) alloc(T * 16 * 4);
  float*  Bvv   = (float*) alloc(T * 16 * 4);
  float*  Cvv   = (float*) alloc(T * 16 * 4);
  float*  dltb  = (float*) alloc(T * DINX * 4);
  __bf16* ybf   = (__bf16*)alloc(T * DINX * 2);
  float*  x1    = (float*) alloc(T * DMODEL * 4);
  __bf16* h2B   = (__bf16*)alloc(T * DMODEL * 2);
  __bf16* mB    = (__bf16*)alloc(T * HIDN * 2);

  // weights -> bf16 (deterministic each call)
  k_cvt_bf16<<<(2 * DINX * DMODEL + 255) / 256, 256, 0, stream>>>(inW,  wInB,  2 * DINX * DMODEL);
  k_cvt_bf16<<<(DMODEL * DINX   + 255) / 256, 256, 0, stream>>>(outW, wOutB, DMODEL * DINX);
  k_cvt_bf16<<<(2 * HIDN * DMODEL + 255) / 256, 256, 0, stream>>>(fc1W, wFc1B, 2 * HIDN * DMODEL);
  k_cvt_bf16<<<(DMODEL * HIDN   + 255) / 256, 256, 0, stream>>>(fc2W, wFc2B, DMODEL * HIDN);

  // --- residual branch 1: mamba mixer ---
  k_rmsnorm<<<T_TOK, 256, 0, stream>>>(x, n1w, xnB);
  k_gemm_inproj<<<dim3(T_TOK / 32, 2), 256, 0, stream>>>(xnB, wInB, ubuf, zbuf);
  k_conv<<<(T_TOK * DINX) / 256, 256, 0, stream>>>(ubuf, convW, convB, ucbuf);
  k_xproj<<<(T_TOK * 48 + 255) / 256, 256, 0, stream>>>(ucbuf, xprojW, dtv, Bvv, Cvv);
  k_dtproj<<<(T_TOK * DINX) / 256, 256, 0, stream>>>(dtv, dtW, dtB, dltb);
  k_scan<<<64, 64, 0, stream>>>(dltb, ucbuf, Bvv, Cvv, zbuf, Alog, Dsk, ybf);
  k_gemm_outproj<<<dim3(T_TOK / 32, 1), 128, 0, stream>>>(ybf, wOutB, x, x1);

  // --- residual branch 2: gated MLP ---
  k_rmsnorm<<<T_TOK, 256, 0, stream>>>(x1, n2w, h2B);
  k_gemm_fc1<<<dim3(T_TOK / 16, 2), 256, 0, stream>>>(h2B, wFc1B, fc1B, mB);
  k_gemm_fc2<<<dim3(T_TOK / 32, 1), 128, 0, stream>>>(mB, wFc2B, fc2B, x1, out);
}
